// SparseMOElayer_29738353557796
// MI455X (gfx1250) — matrix operand order repaired and checked by reading, not compiled, not run on hardware
//
#include <hip/hip_runtime.h>
#include <hip/hip_bf16.h>

// ---------------- problem dims (match reference) ----------------
#define T_TOK  1024   // B*S
#define DMODEL 768
#define NEXP   8
#define HDIM   3072   // 4*D
#define TILE_M 16     // token rows per workgroup (one WMMA M-tile)

typedef __attribute__((ext_vector_type(16))) __bf16 v16bf;
typedef __attribute__((ext_vector_type(8)))  float  v8f;

union Frag { uint4 q[2]; v16bf v; };   // 32B: two aligned 16B loads -> one WMMA operand

// CDNA5 async global->LDS path (guarded: fall back to plain copies if absent)
#if defined(__has_builtin)
#  if __has_builtin(__builtin_amdgcn_global_load_async_to_lds_b128) && \
      __has_builtin(__builtin_amdgcn_s_wait_asynccnt)
#    define HAVE_ASYNC_LDS 1
#  endif
#endif
#ifndef HAVE_ASYNC_LDS
#  define HAVE_ASYNC_LDS 0
#endif

#if HAVE_ASYNC_LDS
// builtin signature (from clang diagnostic): param0 = global 'int __vector(4) *'
typedef int v4i_ __attribute__((vector_size(4 * sizeof(int))));
typedef __attribute__((address_space(1))) v4i_ gv4i;   // global
typedef __attribute__((address_space(3))) v4i_ lv4i;   // LDS
#endif

__device__ __forceinline__ unsigned short f2bf(float f) {
  unsigned int u = __float_as_uint(f);
  u += 0x7fffu + ((u >> 16) & 1u);     // round-to-nearest-even
  return (unsigned short)(u >> 16);
}

__device__ __forceinline__ v8f v8f_zero() {
  v8f z;
  #pragma unroll
  for (int i = 0; i < 8; ++i) z[i] = 0.f;
  return z;
}

// ---------------- fp32 -> bf16 conversion (grid-stride) ----------------
__global__ void cvt_f32_to_bf16(const float* __restrict__ in,
                                unsigned short* __restrict__ out, int n) {
  for (int i = blockIdx.x * blockDim.x + threadIdx.x; i < n; i += gridDim.x * blockDim.x)
    out[i] = f2bf(in[i]);
}

// ---------------- router: logits -> top-2 softmax -> dense gates [T,E] ----------------
// one wave32 per token
__global__ void router_gates(const float* __restrict__ x,
                             const float* __restrict__ Wr,
                             float* __restrict__ gates) {
  const int t    = blockIdx.x;
  const int lane = threadIdx.x;           // blockDim == 32 (wave32)
  float logit[NEXP];
  #pragma unroll
  for (int e = 0; e < NEXP; ++e) {
    float p = 0.f;
    for (int d = lane; d < DMODEL; d += 32) p += x[t * DMODEL + d] * Wr[e * DMODEL + d];
    #pragma unroll
    for (int off = 16; off > 0; off >>= 1) p += __shfl_xor(p, off, 32);
    logit[e] = p;                          // every lane has full reduction
  }
  // top-2 (ties -> lowest index, matching jax.lax.top_k)
  int i0 = 0; float v0 = logit[0];
  #pragma unroll
  for (int e = 1; e < NEXP; ++e) if (logit[e] > v0) { v0 = logit[e]; i0 = e; }
  int i1 = -1; float v1 = -3.0e38f;
  #pragma unroll
  for (int e = 0; e < NEXP; ++e) if (e != i0 && logit[e] > v1) { v1 = logit[e]; i1 = e; }
  const float e1 = __expf(v1 - v0);       // softmax over the two kept logits
  const float g0 = 1.f / (1.f + e1);
  const float g1 = e1  / (1.f + e1);
  if (lane < NEXP) {
    float g = (lane == i0) ? g0 : (lane == i1) ? g1 : 0.f;
    gates[t * NEXP + lane] = g;
  }
}

// ---------------- fused MoE: per 16-token tile, 9 expert passes ----------------
// pass e<8 : acc += gate[t,e] * (relu(x W1[e]^T + b1[e]) W2[e]^T + b2[e])
// pass e==8: acc += 0.2      * (relu(x sW1^T  + sb1 ) sW2^T  + sb2 )   (shared, 0.1*K)
__global__ void __launch_bounds__(256)
moe_fused(const unsigned short* __restrict__ xbf,
          const unsigned short* __restrict__ W1bf, const float* __restrict__ b1,
          const unsigned short* __restrict__ W2bf, const float* __restrict__ b2,
          const unsigned short* __restrict__ sW1bf, const float* __restrict__ sb1,
          const unsigned short* __restrict__ sW2bf, const float* __restrict__ sb2,
          const float* __restrict__ gates,
          float* __restrict__ out) {
  extern __shared__ unsigned char smem_raw[];
  unsigned short* x_s = (unsigned short*)smem_raw;          // [16][768]  bf16 : 24 KB
  unsigned short* h_s = x_s + TILE_M * DMODEL;              // [16][3072] bf16 : 96 KB
  float*          g_s = (float*)(h_s + TILE_M * HDIM);      // [16][8]    f32  : 512 B

  const int tid  = threadIdx.x;
  const int wave = tid >> 5;       // 0..7
  const int lane = tid & 31;
  const int half = lane >> 4;      // 0/1 : which 16-lane half
  const int l15  = lane & 15;
  const int t0   = blockIdx.x * TILE_M;

  // ---- stage x tile (contiguous 24KB region of xbf) + gate tile into LDS ----
  const unsigned short* gsrc = xbf + (size_t)t0 * DMODEL;
#if HAVE_ASYNC_LDS
  {
    gv4i* gp = (gv4i*)(const void*)gsrc;
    lv4i* lp = (lv4i*)(void*)x_s;
    for (int c = tid; c < (TILE_M * DMODEL) / 8; c += blockDim.x)   // 16B chunks
      __builtin_amdgcn_global_load_async_to_lds_b128(gp + c, lp + c, 0, 0);
  }
#else
  for (int c = tid; c < (TILE_M * DMODEL) / 8; c += blockDim.x)
    ((uint4*)x_s)[c] = ((const uint4*)gsrc)[c];
#endif
  for (int i = tid; i < TILE_M * NEXP; i += blockDim.x)
    g_s[i] = gates[(size_t)t0 * NEXP + i];
#if HAVE_ASYNC_LDS
  __builtin_amdgcn_s_wait_asynccnt(0);
#endif
  __syncthreads();

  // persistent f32 output accumulators: wave owns N-tiles {wave, wave+8, ..., wave+40}
  v8f acc[6];
  #pragma unroll
  for (int ti = 0; ti < 6; ++ti) acc[ti] = v8f_zero();

  for (int e = 0; e <= NEXP; ++e) {
    const bool se = (e == NEXP);
    const unsigned short* Wa = se ? sW1bf : (W1bf + (size_t)e * HDIM * DMODEL);
    const float*          ba = se ? sb1   : (b1 + e * HDIM);
    const unsigned short* Wb = se ? sW2bf : (W2bf + (size_t)e * DMODEL * HDIM);
    const float*          bb = se ? sb2   : (b2 + e * DMODEL);

    // ---- phase 1: h[16][HDIM] = relu(x @ Wa^T + ba) -> LDS (bf16) ----
    // wave owns 24 of 192 N-tiles; process 4 at a time sharing one A fragment
    // => 4 independent WMMA chains per A load (ILP + 4x less LDS traffic)
    const unsigned short* arow1 = x_s + l15 * DMODEL + half * 8;
    for (int grp = 0; grp < 6; ++grp) {
      int n[4];
      const unsigned short* brow[4];
      v8f c[4];
      #pragma unroll
      for (int t = 0; t < 4; ++t) {
        const int nt = wave + grp * 32 + t * 8;               // uniform per wave
        n[t]    = nt * 16 + l15;
        brow[t] = Wa + (size_t)n[t] * DMODEL + half * 8;
        c[t]    = v8f_zero();
        if (grp + 1 < 6)                                      // prefetch next group's row
          __builtin_prefetch((const void*)(brow[t] + (size_t)512 * DMODEL), 0, 1);
      }
      #pragma unroll 2
      for (int k0 = 0; k0 < DMODEL / 32; ++k0) {
        Frag a;
        a.q[0] = *(const uint4*)(arow1 + k0 * 32);            // K = k0*32 + half*8 + [0..7]
        a.q[1] = *(const uint4*)(arow1 + k0 * 32 + 16);       // K += 16
        #pragma unroll
        for (int t = 0; t < 4; ++t) {
          Frag b;
          b.q[0] = *(const uint4*)(brow[t] + k0 * 32);        // K = k0*32 + half*8 + [0..7]
          b.q[1] = *(const uint4*)(brow[t] + k0 * 32 + 16);   // K += 16
          c[t] = __builtin_amdgcn_wmma_f32_16x16x32_bf16(false, a.v, false, b.v,
                                                         (short)0, c[t], false, false);
        }
      }
      #pragma unroll
      for (int t = 0; t < 4; ++t) {
        const float bn = ba[n[t]];
        #pragma unroll
        for (int j = 0; j < 8; ++j) {                         // C: row j+8*half, col l15
          const int m = j + half * 8;
          h_s[m * HDIM + n[t]] = f2bf(fmaxf(c[t][j] + bn, 0.f));
        }
      }
    }
    __syncthreads();

    // ---- phase 2: acc += gate * (h @ Wb^T + bb) ----
    // A fragment is IDENTICAL for all 6 N-tiles of this wave: load once per k-step,
    // feed 6 independent WMMA chains.
    {
      int nn[6];
      const unsigned short* brow2[6];
      v8f c2[6];
      #pragma unroll
      for (int ti = 0; ti < 6; ++ti) {
        const int nt = wave + ti * 8;                         // 48 N-tiles over DMODEL
        nn[ti]    = nt * 16 + l15;
        brow2[ti] = Wb + (size_t)nn[ti] * HDIM + half * 8;
        c2[ti]    = v8f_zero();
        __builtin_prefetch((const void*)brow2[ti], 0, 1);
      }
      const unsigned short* arow2 = h_s + l15 * HDIM + half * 8;
      #pragma unroll 2
      for (int k0 = 0; k0 < HDIM / 32; ++k0) {
        Frag a;
        a.q[0] = *(const uint4*)(arow2 + k0 * 32);
        a.q[1] = *(const uint4*)(arow2 + k0 * 32 + 16);
        #pragma unroll
        for (int ti = 0; ti < 6; ++ti) {
          Frag b;
          b.q[0] = *(const uint4*)(brow2[ti] + k0 * 32);
          b.q[1] = *(const uint4*)(brow2[ti] + k0 * 32 + 16);
          c2[ti] = __builtin_amdgcn_wmma_f32_16x16x32_bf16(false, a.v, false, b.v,
                                                           (short)0, c2[ti], false, false);
        }
      }
      #pragma unroll
      for (int ti = 0; ti < 6; ++ti) {
        const float bn = bb[nn[ti]];
        #pragma unroll
        for (int j = 0; j < 8; ++j) {
          const int m = j + half * 8;
          const float g = se ? 0.2f : g_s[m * NEXP + e];
          acc[ti][j] += g * (c2[ti][j] + bn);
        }
      }
    }
    __syncthreads();   // h_s reused next pass
  }

  // ---- single write of the finished 16xD f32 tile ----
  #pragma unroll
  for (int ti = 0; ti < 6; ++ti) {
    const int nt = wave + ti * 8;
    const int n  = nt * 16 + l15;
    #pragma unroll
    for (int j = 0; j < 8; ++j) {
      const int m = j + half * 8;
      out[(size_t)(t0 + m) * DMODEL + n] = acc[ti][j];
    }
  }
}

// ---------------- host launcher ----------------
extern "C" void kernel_launch(void* const* d_in, const int* in_sizes, int n_in,
                              void* d_out, int out_size, void* d_ws, size_t ws_size,
                              hipStream_t stream) {
  const float* x   = (const float*)d_in[0];
  const float* Wr  = (const float*)d_in[1];
  const float* W1  = (const float*)d_in[2];
  const float* b1  = (const float*)d_in[3];
  const float* W2  = (const float*)d_in[4];
  const float* b2  = (const float*)d_in[5];
  const float* sW1 = (const float*)d_in[6];
  const float* sb1 = (const float*)d_in[7];
  const float* sW2 = (const float*)d_in[8];
  const float* sb2 = (const float*)d_in[9];
  float* out = (float*)d_out;
  (void)in_sizes; (void)n_in; (void)out_size; (void)ws_size;

  // workspace carve-out (~86.6 MB), 256B aligned slices
  char* ws = (char*)d_ws;
  size_t off = 0;
  auto carve = [&](size_t bytes) -> char* {
    char* p = ws + off;
    off = (off + bytes + 255) & ~(size_t)255;
    return p;
  };
  float*          gates = (float*)         carve((size_t)T_TOK * NEXP * 4);
  unsigned short* xbf   = (unsigned short*)carve((size_t)T_TOK * DMODEL * 2);
  unsigned short* W1bf  = (unsigned short*)carve((size_t)NEXP * HDIM * DMODEL * 2);
  unsigned short* W2bf  = (unsigned short*)carve((size_t)NEXP * DMODEL * HDIM * 2);
  unsigned short* sW1bf = (unsigned short*)carve((size_t)HDIM * DMODEL * 2);
  unsigned short* sW2bf = (unsigned short*)carve((size_t)DMODEL * HDIM * 2);

  // fp32 -> bf16 staging (streams ~170MB; bf16 weights (75MB) then live in 192MB L2)
  cvt_f32_to_bf16<<<1024, 256, 0, stream>>>(x,   xbf,   T_TOK * DMODEL);
  cvt_f32_to_bf16<<<4096, 256, 0, stream>>>(W1,  W1bf,  NEXP * HDIM * DMODEL);
  cvt_f32_to_bf16<<<4096, 256, 0, stream>>>(W2,  W2bf,  NEXP * DMODEL * HDIM);
  cvt_f32_to_bf16<<<2048, 256, 0, stream>>>(sW1, sW1bf, HDIM * DMODEL);
  cvt_f32_to_bf16<<<2048, 256, 0, stream>>>(sW2, sW2bf, DMODEL * HDIM);

  router_gates<<<T_TOK, 32, 0, stream>>>(x, Wr, gates);

  const size_t smem = (size_t)TILE_M * DMODEL * 2   // x tile  (24 KB)
                    + (size_t)TILE_M * HDIM * 2     // h tile  (96 KB)  -- CDNA5 320KB LDS
                    + (size_t)TILE_M * NEXP * 4;    // gates   (512 B)
  (void)hipFuncSetAttribute(reinterpret_cast<const void*>(moe_fused),
                            hipFuncAttributeMaxDynamicSharedMemorySize, (int)smem);
  moe_fused<<<T_TOK / TILE_M, 256, smem, stream>>>(xbf, W1bf, b1, W2bf, b2,
                                                   sW1bf, sb1, sW2bf, sb2, gates, out);
}
